// MoELayer_39152921870918
// MI455X (gfx1250) — matrix operand order repaired and checked by reading, not compiled, not run on hardware
//
#include <hip/hip_runtime.h>
#include <hip/hip_bf16.h>

// ---------------------------------------------------------------------------
// MoE layer for MI455X (gfx1250, wave32, WMMA + async global->LDS).
// DIM=128, E=8, K(topk)=2, F*OUT=480, H=512, B=64, A_TOT=129 -> N=128 tokens.
// d_out layout (flat, return order): output[64*128*480] | probs[64*128*8] |
//                                    all_expert_outputs[64*128*8*480]
// Workspace: top2 idx (64KB) | top2 p (64KB) | h bf16 (64MB) | xb bf16 (2MB) |
//            w1t bf16 (1MB) | w2t bf16 (3.75MB)
// ---------------------------------------------------------------------------

#define DIM   128
#define NEXP  8
#define HDIM  512
#define FOUT  480
#define NTOK  128
#define NBAT  64
#define ATOT  129

typedef __attribute__((ext_vector_type(16))) __bf16 v16bf;
typedef __attribute__((ext_vector_type(8)))  float  v8f;

// ---- CDNA5 async global->LDS copy (ASYNCcnt-tracked, bypasses VGPRs) ------
// VGLOBAL encoding: VDST = per-lane LDS byte offset, VADDR = 64-bit address.
__device__ __forceinline__ void async_copy_b128(unsigned lds_off, const void* gptr) {
  asm volatile("global_load_async_to_lds_b128 %0, %1, off"
               :: "v"(lds_off), "v"((unsigned long long)(size_t)gptr)
               : "memory");
}
__device__ __forceinline__ void async_wait0() {
  asm volatile("s_wait_asynccnt 0" ::: "memory");
}
// Flat shared-pointer -> LDS byte offset (LDS aperture: low 32 bits of the address).
__device__ __forceinline__ unsigned lds_offset(const void* p) {
  return (unsigned)(size_t)p;
}

// ---- WMMA fragment loaders (ISA 7.12.2 layouts, wave32) -------------------

// A-matrix 16x32 bf16: lane l holds row m0 + (l%16); K values
// k0 + g*8 + {0..7} (elems 0..7) and k0 + 16 + g*8 + {0..7} (elems 8..15).
__device__ __forceinline__ v16bf load_A_frag(const __bf16* base, int lda,
                                             int m0, int k0, int lane) {
  const __bf16* p = base + (size_t)(m0 + (lane & 15)) * lda + k0 + ((lane >> 4) << 3);
  union { uint4 u[2]; v16bf v; } r;
  r.u[0] = *(const uint4*)p;        // K: +0..+7
  r.u[1] = *(const uint4*)(p + 16); // K: +16..+23
  return r.v;
}

// B-matrix 32x16 bf16: lane l holds column n0 + (l%16); K values
// k0 + (l/16)*8 + {0..7} (elems 0..7) and k0 + 16 + (l/16)*8 + {0..7} (elems 8..15). Source is LDS tile stored [n][k].
__device__ __forceinline__ v16bf load_B_frag(const __bf16* baseT, int ldb,
                                             int n0, int k0, int lane) {
  const __bf16* p = baseT + (size_t)(n0 + (lane & 15)) * ldb + k0 + ((lane >> 4) << 3);
  union { uint4 u[2]; v16bf v; } r;
  r.u[0] = *(const uint4*)p;
  r.u[1] = *(const uint4*)(p + 16);
  return r.v;
}

// ---- Kernel 0: one-time fp32 -> bf16 converts (+ B-layout transposes) -----
__global__ void convert_x_kernel(const float* __restrict__ x, __bf16* __restrict__ xb) {
  int i = blockIdx.x * blockDim.x + threadIdx.x;       // 64*128*128
  if (i >= NBAT * NTOK * DIM) return;
  int d = i & 127, rest = i >> 7, n = rest & 127, b = rest >> 7;
  xb[i] = (__bf16)x[((size_t)b * ATOT + 1 + n) * DIM + d];
}
__global__ void convert_w1_kernel(const float* __restrict__ w1, __bf16* __restrict__ w1t) {
  int i = blockIdx.x * blockDim.x + threadIdx.x;       // 8*128*512 (source order)
  if (i >= NEXP * DIM * HDIM) return;
  int h = i & 511, rest = i >> 9, k = rest & 127, e = rest >> 7;
  w1t[((size_t)e * HDIM + h) * DIM + k] = (__bf16)w1[i];   // [e][h][k]
}
__global__ void convert_w2_kernel(const float* __restrict__ w2, __bf16* __restrict__ w2t) {
  int i = blockIdx.x * blockDim.x + threadIdx.x;       // 8*512*480 (source order)
  if (i >= NEXP * HDIM * FOUT) return;
  int o = i % FOUT, rest = i / FOUT, k = rest & 511, e = rest >> 9;
  w2t[((size_t)e * FOUT + o) * HDIM + k] = (__bf16)w2[i];  // [e][o][k]
}

// ---- Kernel 1: router (logits, softmax -> probs, top-2) -------------------
__global__ void router_kernel(const float* __restrict__ x,
                              const float* __restrict__ rw,
                              const float* __restrict__ rb,
                              float* __restrict__ probs_out,
                              int* __restrict__ top_idx,
                              float* __restrict__ top_p) {
  int t = blockIdx.x * blockDim.x + threadIdx.x;       // token id 0..8191
  if (t >= NBAT * NTOK) return;
  int b = t / NTOK, n = t % NTOK;
  const float* xr = x + ((size_t)b * ATOT + 1 + n) * DIM;

  float logits[NEXP];
#pragma unroll
  for (int e = 0; e < NEXP; ++e) logits[e] = rb[e];
  for (int d = 0; d < DIM; ++d) {
    float xv = xr[d];
#pragma unroll
    for (int e = 0; e < NEXP; ++e) logits[e] += xv * rw[d * NEXP + e];
  }
  float mx = logits[0];
#pragma unroll
  for (int e = 1; e < NEXP; ++e) mx = fmaxf(mx, logits[e]);
  float sum = 0.f, p[NEXP];
#pragma unroll
  for (int e = 0; e < NEXP; ++e) { p[e] = __expf(logits[e] - mx); sum += p[e]; }
  float inv = 1.f / sum;
#pragma unroll
  for (int e = 0; e < NEXP; ++e) { p[e] *= inv; probs_out[(size_t)t * NEXP + e] = p[e]; }

  int   i0 = 0; float p0 = p[0];
#pragma unroll
  for (int e = 1; e < NEXP; ++e) if (p[e] > p0) { p0 = p[e]; i0 = e; }
  int   i1 = (i0 == 0) ? 1 : 0; float p1 = p[i1];
#pragma unroll
  for (int e = 0; e < NEXP; ++e) if (e != i0 && p[e] > p1) { p1 = p[e]; i1 = e; }
  top_idx[t * 2 + 0] = i0; top_idx[t * 2 + 1] = i1;
  top_p[t * 2 + 0] = p0;   top_p[t * 2 + 1] = p1;
}

// ---- Kernel 2: h = relu(X @ W1 + b1), bf16 out to workspace ---------------
// grid = (HDIM/128, NEXP, NBAT), block = 128 threads (4 waves).
// M=128 tokens x N=128 h-cols, K=128. Tiles staged via async global->LDS.
__global__ void __launch_bounds__(128)
moe_gemm1_kernel(const __bf16* __restrict__ xb,
                 const __bf16* __restrict__ w1t,
                 const float* __restrict__ b1,
                 __bf16* __restrict__ h_ws) {
  __shared__ __bf16 sX[NTOK * DIM];   // [m][k]   32 KB
  __shared__ __bf16 sW[128 * DIM];    // [n][k]   32 KB (B-layout tile)

  const int tid = threadIdx.x;
  const int h0 = blockIdx.x * 128;
  const int e  = blockIdx.y;
  const int b  = blockIdx.z;

  // Both tiles are contiguous 32KB blocks in bf16 workspace -> flat async copy.
  const __bf16* xg = xb  + (size_t)b * NTOK * DIM;
  const __bf16* wg = w1t + ((size_t)e * HDIM + h0) * DIM;
  const unsigned sX_off = lds_offset(sX), sW_off = lds_offset(sW);
#pragma unroll 4
  for (int i = tid; i < NTOK * DIM / 8; i += 128) {      // 2048 x 16B chunks
    async_copy_b128(sX_off + i * 16, xg + i * 8);
    async_copy_b128(sW_off + i * 16, wg + i * 8);
  }
  async_wait0();
  __syncthreads();

  const int lane = tid & 31, w = tid >> 5;
  const int rowOff = (lane >> 4) << 3;  // C/D: M = v + rowOff
  const int col    = lane & 15;         // C/D: N = col

  v8f acc[2][8] = {};                   // wave: M rows [w*32, w*32+32), N 0..127
#pragma unroll
  for (int kc = 0; kc < DIM; kc += 32) {
    v16bf a0 = load_A_frag(sX, DIM, w * 32 +  0, kc, lane);
    v16bf a1 = load_A_frag(sX, DIM, w * 32 + 16, kc, lane);
#pragma unroll
    for (int nt = 0; nt < 8; ++nt) {
      v16bf bf = load_B_frag(sW, DIM, nt * 16, kc, lane);
      acc[0][nt] = __builtin_amdgcn_wmma_f32_16x16x32_bf16(
          false, a0, false, bf, (short)0, acc[0][nt], false, false);
      acc[1][nt] = __builtin_amdgcn_wmma_f32_16x16x32_bf16(
          false, a1, false, bf, (short)0, acc[1][nt], false, false);
    }
  }

  // Epilogue: +bias, relu, bf16 store to h_ws[b][e][m][h]
#pragma unroll
  for (int mt = 0; mt < 2; ++mt) {
#pragma unroll
    for (int nt = 0; nt < 8; ++nt) {
      float bias = b1[(size_t)e * HDIM + h0 + nt * 16 + col];
#pragma unroll
      for (int v = 0; v < 8; ++v) {
        int m = w * 32 + mt * 16 + rowOff + v;
        float hv = acc[mt][nt][v] + bias;
        hv = hv > 0.f ? hv : 0.f;
        h_ws[(((size_t)b * NEXP + e) * NTOK + m) * HDIM + h0 + nt * 16 + col] = (__bf16)hv;
      }
    }
  }
}

// ---- Kernel 3: y = h @ W2 + b2 -> all_expert_outputs ----------------------
// grid = (FOUT/96, NEXP, NBAT), block = 256 threads (8 waves).
// M=128 tokens x N=96 out-cols, K=512 staged in 64-wide async LDS chunks.
__global__ void __launch_bounds__(256)
moe_gemm2_kernel(const __bf16* __restrict__ h_ws,
                 const __bf16* __restrict__ w2t,
                 const float* __restrict__ b2,
                 float* __restrict__ aeo) {
  __shared__ __bf16 sH[NTOK * 64];    // [m][kk]  16 KB
  __shared__ __bf16 sW2[96 * 64];     // [n][kk]  12 KB

  const int tid = threadIdx.x;
  const int j0 = blockIdx.x * 96;
  const int e  = blockIdx.y;
  const int b  = blockIdx.z;

  const int lane = tid & 31, w = tid >> 5;
  const int rowOff = (lane >> 4) << 3;
  const int col    = lane & 15;

  const __bf16* hg  = h_ws + ((size_t)b * NEXP + e) * NTOK * HDIM;  // [m][512]
  const __bf16* w2g = w2t + ((size_t)e * FOUT + j0) * HDIM;         // [n][512]
  const unsigned sH_off = lds_offset(sH), sW2_off = lds_offset(sW2);

  v8f acc[6] = {};                    // wave: M rows [w*16, w*16+16), N 0..95
  for (int kc = 0; kc < HDIM; kc += 64) {
    // Hint next h chunk toward L2 while this one computes (global_prefetch_b8).
    if (kc + 64 < HDIM)
      __builtin_prefetch(hg + (size_t)(tid >> 1) * HDIM + kc + 64, 0, 1);

#pragma unroll
    for (int i = tid; i < NTOK * 64 / 8; i += 256) {     // 1024 x 16B chunks
      int m = i >> 3, kq = (i & 7) << 3;
      async_copy_b128(sH_off + i * 16, hg + (size_t)m * HDIM + kc + kq);
    }
#pragma unroll
    for (int i = tid; i < 96 * 64 / 8; i += 256) {       // 768 x 16B chunks
      int n = i >> 3, kq = (i & 7) << 3;
      async_copy_b128(sW2_off + i * 16, w2g + (size_t)n * HDIM + kc + kq);
    }
    async_wait0();
    __syncthreads();

#pragma unroll
    for (int k0 = 0; k0 < 64; k0 += 32) {
      v16bf a = load_A_frag(sH, 64, w * 16, k0, lane);
#pragma unroll
      for (int nt = 0; nt < 6; ++nt) {
        v16bf bf = load_B_frag(sW2, 64, nt * 16, k0, lane);
        acc[nt] = __builtin_amdgcn_wmma_f32_16x16x32_bf16(
            false, a, false, bf, (short)0, acc[nt], false, false);
      }
    }
    __syncthreads();
  }

  // Epilogue: +b2, write to all_expert_outputs[b][m][e][j] (reference layout).
#pragma unroll
  for (int nt = 0; nt < 6; ++nt) {
    float bias = b2[(size_t)e * FOUT + j0 + nt * 16 + col];
#pragma unroll
    for (int v = 0; v < 8; ++v) {
      int m = w * 16 + rowOff + v;
      int n = j0 + nt * 16 + col;
      aeo[(((size_t)b * NTOK + m) * NEXP + e) * FOUT + n] = acc[nt][v] + bias;
    }
  }
}

// ---- Kernel 4: output = sum_k top_p * aeo[., top_idx, .] ------------------
__global__ void combine_kernel(const float* __restrict__ aeo,
                               const int* __restrict__ top_idx,
                               const float* __restrict__ top_p,
                               float* __restrict__ out) {
  int t = blockIdx.x;                              // token id 0..8191
  int i0 = top_idx[t * 2 + 0], i1 = top_idx[t * 2 + 1];
  float p0 = top_p[t * 2 + 0], p1 = top_p[t * 2 + 1];
  const float* r0 = aeo + ((size_t)t * NEXP + i0) * FOUT;
  const float* r1 = aeo + ((size_t)t * NEXP + i1) * FOUT;
  float* o = out + (size_t)t * FOUT;
  for (int j = threadIdx.x; j < FOUT; j += blockDim.x)
    o[j] = p0 * r0[j] + p1 * r1[j];
}

// ---------------------------------------------------------------------------
extern "C" void kernel_launch(void* const* d_in, const int* in_sizes, int n_in,
                              void* d_out, int out_size, void* d_ws, size_t ws_size,
                              hipStream_t stream) {
  const float* x        = (const float*)d_in[0];
  // d_in[1] = A (scalar 129), compile-time constant here
  const float* router_w = (const float*)d_in[2];
  const float* router_b = (const float*)d_in[3];
  const float* w1       = (const float*)d_in[4];
  const float* b1       = (const float*)d_in[5];
  const float* w2       = (const float*)d_in[6];
  const float* b2       = (const float*)d_in[7];

  float* out_ptr   = (float*)d_out;                                  // 64*128*480
  float* probs_ptr = out_ptr + (size_t)NBAT * NTOK * FOUT;           // 64*128*8
  float* aeo_ptr   = probs_ptr + (size_t)NBAT * NTOK * NEXP;         // 64*128*8*480

  char* ws = (char*)d_ws;
  int*    top_idx = (int*)ws;                                        // 64 KB
  float*  top_p   = (float*)(ws + (size_t)64 * 1024);                // 64 KB
  __bf16* h_ws    = (__bf16*)(ws + (size_t)128 * 1024);              // 64 MB
  __bf16* xb      = (__bf16*)(ws + (size_t)128 * 1024 + (size_t)64 * 1024 * 1024);
  __bf16* w1t     = xb  + (size_t)NBAT * NTOK * DIM;                 // 2 MB after xb
  __bf16* w2t     = w1t + (size_t)NEXP * HDIM * DIM;                 // 1 MB after w1t

  convert_x_kernel<<<(NBAT * NTOK * DIM + 255) / 256, 256, 0, stream>>>(x, xb);
  convert_w1_kernel<<<(NEXP * DIM * HDIM + 255) / 256, 256, 0, stream>>>(w1, w1t);
  convert_w2_kernel<<<(NEXP * HDIM * FOUT + 255) / 256, 256, 0, stream>>>(w2, w2t);

  router_kernel<<<(NBAT * NTOK + 255) / 256, 256, 0, stream>>>(
      x, router_w, router_b, probs_ptr, top_idx, top_p);

  moe_gemm1_kernel<<<dim3(HDIM / 128, NEXP, NBAT), 128, 0, stream>>>(
      xb, w1t, b1, h_ws);

  moe_gemm2_kernel<<<dim3(FOUT / 96, NEXP, NBAT), 256, 0, stream>>>(
      h_ws, w2t, b2, aeo_ptr);

  combine_kernel<<<NBAT * NTOK, 128, 0, stream>>>(
      aeo_ptr, top_idx, top_p, out_ptr);
}
